// GATLayer_3255585210887
// MI455X (gfx1250) — compile-verified, matrix operand order checked by reading, not run on hardware
//
#include <hip/hip_runtime.h>
#include <hip/hip_bf16.h>

typedef __attribute__((ext_vector_type(16))) __bf16 v16bf;
typedef __attribute__((ext_vector_type(8)))  __bf16 v8bf;
typedef __attribute__((ext_vector_type(8)))  float  v8f;
typedef __attribute__((ext_vector_type(4)))  float  v4f;
typedef __attribute__((ext_vector_type(4)))  int    v4i;

#define N_NODES 8192
#define FIN     128
#define FOUT    64
#define ALPHA   0.2f
#define SPLIT   4                      // j-dimension splits for occupancy
#define JCHUNK  (N_NODES / SPLIT)      // 2048 columns per split

// ---------------------------------------------------------------------------
// Kernel 1: h = inp @ W  via bf16 WMMA.
//  - writes hF  : h as plain f32 row-major (for exact f1/f2 computation)
//  - writes hB  : h as bf16 packed in WMMA B-matrix lane layout:
//      element (j, f) -> tile (jt=j>>5, ft=f>>4),
//      lane  = (f&15) + 16*((j_local>>3)&1)
//      pos   = ((j_local>>4)<<3) | (j_local&7)
//      index = (jt*4+ft)*512 + lane*16 + pos
// ---------------------------------------------------------------------------
__global__ __launch_bounds__(256) void k_h_gemm(const float* __restrict__ inp,
                                                const float* __restrict__ W,
                                                float*  __restrict__ hF,
                                                __bf16* __restrict__ hB)
{
    __shared__ __bf16 ldsW[16 * 512];   // 16 (kt,ft) tiles in B-layout, 16KB
    const int tid = threadIdx.x;

    // cooperatively convert W (128x64 f32) into bf16 B-layout tiles in LDS
    for (int idx = tid; idx < FIN * FOUT; idx += 256) {
        int k  = idx >> 6;            // row of W (Fin)
        int f  = idx & 63;            // col of W (Fout)
        int kt = k >> 5, ft = f >> 4;
        int kl = k & 31;
        int hl = (kl >> 3) & 1;
        int pos = ((kl >> 4) << 3) | (kl & 7);
        int lane = (f & 15) + (hl << 4);
        ldsW[((kt << 2) + ft) * 512 + lane * 16 + pos] = (__bf16)W[idx];
    }
    __syncthreads();

    const int wave = tid >> 5;
    const int lane = tid & 31;
    const int hl   = lane >> 4;       // lane half
    const int n    = lane & 15;
    const int tile = blockIdx.x * 8 + wave;   // 64 blocks * 8 waves = 512 tiles
    const int m0   = tile * 16;
    const int row  = m0 + n;          // A-matrix: lane holds M = lane&15

    v8f acc[4] = {};
    const float* arow = inp + (size_t)row * FIN;

    #pragma unroll
    for (int kt = 0; kt < 4; ++kt) {
        const int k0 = kt * 32 + hl * 8;
        v4f a0 = *(const v4f*)(arow + k0);
        v4f a1 = *(const v4f*)(arow + k0 + 4);
        v4f a2 = *(const v4f*)(arow + k0 + 16);
        v4f a3 = *(const v4f*)(arow + k0 + 20);
        v16bf av;
        #pragma unroll
        for (int t = 0; t < 4; ++t) {
            av[t]      = (__bf16)a0[t];
            av[4 + t]  = (__bf16)a1[t];
            av[8 + t]  = (__bf16)a2[t];
            av[12 + t] = (__bf16)a3[t];
        }
        #pragma unroll
        for (int ft = 0; ft < 4; ++ft) {
            v16bf bv = *(const v16bf*)&ldsW[((kt << 2) + ft) * 512 + lane * 16];
            acc[ft] = __builtin_amdgcn_wmma_f32_16x16x32_bf16(
                false, av, false, bv, (short)0, acc[ft], false, false);
        }
    }

    // C layout: element v of acc = row (m0 + v + 8*hl), col (ft*16 + n)
    const int jt   = m0 >> 5;
    const int pos0 = (m0 & 16) >> 1;   // 0 or 8 (which K-group of the 32-tile)
    #pragma unroll
    for (int ft = 0; ft < 4; ++ft) {
        v8bf pk;
        #pragma unroll
        for (int v = 0; v < 8; ++v) {
            float x = acc[ft][v];
            pk[v] = (__bf16)x;
            hF[(size_t)(m0 + v + hl * 8) * FOUT + ft * 16 + n] = x;
        }
        *(v8bf*)&hB[(size_t)((jt * 4 + ft) * 512 + lane * 16 + pos0)] = pk;
    }
}

// ---------------------------------------------------------------------------
// Kernel 2: f1 = h@a1, f2 = h@a2   (one thread per row)
// ---------------------------------------------------------------------------
__global__ __launch_bounds__(256) void k_f1f2(const float* __restrict__ hF,
                                              const float* __restrict__ a,
                                              float* __restrict__ f1,
                                              float* __restrict__ f2)
{
    int r = blockIdx.x * 256 + threadIdx.x;
    const float* hr = hF + (size_t)r * FOUT;
    float s1 = 0.f, s2 = 0.f;
    #pragma unroll
    for (int f = 0; f < FOUT; f += 4) {
        v4f hv = *(const v4f*)(hr + f);
        v4f a1 = *(const v4f*)(a + f);
        v4f a2 = *(const v4f*)(a + FOUT + f);
        #pragma unroll
        for (int t = 0; t < 4; ++t) { s1 += hv[t] * a1[t]; s2 += hv[t] * a2[t]; }
    }
    f1[r] = s1;
    f2[r] = s2;
}

// ---------------------------------------------------------------------------
// Kernel 3: global max of f2 (safe softmax shift bound)
// ---------------------------------------------------------------------------
__global__ __launch_bounds__(256) void k_maxf2(const float* __restrict__ f2,
                                               float* __restrict__ mx)
{
    __shared__ float red[256];
    int tid = threadIdx.x;
    float m = -3.0e38f;
    for (int i = tid; i < N_NODES; i += 256) m = fmaxf(m, f2[i]);
    red[tid] = m;
    __syncthreads();
    for (int s = 128; s > 0; s >>= 1) {
        if (tid < s) red[tid] = fmaxf(red[tid], red[tid + s]);
        __syncthreads();
    }
    if (tid == 0) mx[0] = red[0];
}

// ---------------------------------------------------------------------------
// Kernel 4: fused masked softmax + attention@h (bf16 WMMA), split over j.
// Grid: 64*SPLIT blocks of 8 waves; wave = one 16-row tile x one j-chunk.
// m_i = lrelu(f1[i] + max f2) >= true row max -> exp(e - m_i) <= 1, so the
// splits share one shift and partials can be summed without rescaling.
// Writes: pacc[split][row][f] (un-normalized), psum[split][row].
// ---------------------------------------------------------------------------
__global__ __launch_bounds__(256) void k_gat_main(const int*    __restrict__ adj,
                                                  const __bf16* __restrict__ hB,
                                                  const float*  __restrict__ f1,
                                                  const float*  __restrict__ f2,
                                                  const float*  __restrict__ mx,
                                                  float*        __restrict__ pacc,
                                                  float*        __restrict__ psum)
{
    __shared__ float lf2[JCHUNK];       // 8KB: this split's slice of f2
    const int tid   = threadIdx.x;
    const int split = blockIdx.x >> 6;  // 0..SPLIT-1
    const int jbeg  = split * JCHUNK;

    for (int i = tid; i < JCHUNK; i += 256) lf2[i] = f2[jbeg + i];
    __syncthreads();

    const int wave = tid >> 5;
    const int lane = tid & 31;
    const int hl   = lane >> 4;
    const int n    = lane & 15;
    const int tile = (blockIdx.x & 63) * 8 + wave;
    const int m0   = tile * 16;

    const float f1v  = f1[m0 + n];            // A-matrix row of this lane
    const float tmax = f1v + mx[0];
    const float mi   = fmaxf(tmax, ALPHA * tmax);   // lrelu(f1 + maxF2)

    // running pointers: avoid per-iteration 64-bit address recompute
    const int*    aptr  = adj + (size_t)(m0 + n) * N_NODES + jbeg + hl * 8;
    const __bf16* hptr  = hB  + (size_t)(jbeg >> 5) * 2048 + lane * 16;
    const float*  f2ptr = &lf2[hl * 8];

    v8f   acc[4] = {};
    float ps = 0.f;                            // partial row sum (this lane-half)

    #pragma unroll 2
    for (int it = 0; it < JCHUNK / 32; ++it) {
        // streaming non-temporal adjacency loads (read-once, 256MB total)
        v4i mA0 = __builtin_nontemporal_load((const v4i*)(aptr));
        v4i mA1 = __builtin_nontemporal_load((const v4i*)(aptr + 4));
        v4i mB0 = __builtin_nontemporal_load((const v4i*)(aptr + 16));
        v4i mB1 = __builtin_nontemporal_load((const v4i*)(aptr + 20));
        v4f fA0 = *(const v4f*)(f2ptr);
        v4f fA1 = *(const v4f*)(f2ptr + 4);
        v4f fB0 = *(const v4f*)(f2ptr + 16);
        v4f fB1 = *(const v4f*)(f2ptr + 20);

        v16bf av;
        #pragma unroll
        for (int t = 0; t < 4; ++t) {
            float x, l, p;
            x = f1v + fA0[t]; l = fmaxf(x, ALPHA * x);
            p = (mA0[t] > 0) ? __expf(l - mi) : 0.f;  ps += p; av[t]      = (__bf16)p;
            x = f1v + fA1[t]; l = fmaxf(x, ALPHA * x);
            p = (mA1[t] > 0) ? __expf(l - mi) : 0.f;  ps += p; av[4 + t]  = (__bf16)p;
            x = f1v + fB0[t]; l = fmaxf(x, ALPHA * x);
            p = (mB0[t] > 0) ? __expf(l - mi) : 0.f;  ps += p; av[8 + t]  = (__bf16)p;
            x = f1v + fB1[t]; l = fmaxf(x, ALPHA * x);
            p = (mB1[t] > 0) ? __expf(l - mi) : 0.f;  ps += p; av[12 + t] = (__bf16)p;
        }

        #pragma unroll
        for (int ft = 0; ft < 4; ++ft) {
            v16bf bv = *(const v16bf*)(hptr + ft * 512);
            acc[ft] = __builtin_amdgcn_wmma_f32_16x16x32_bf16(
                false, av, false, bv, (short)0, acc[ft], false, false);
        }

        aptr  += 32;
        f2ptr += 32;
        hptr  += 2048;                 // next 32-j tile group (4 f-tiles * 512)
    }

    // full row sums: lane l and l+16 hold complementary halves of row m0+(l&15)
    float full = ps + __shfl_xor(ps, 16, 32);
    if (lane < 16) psum[split * N_NODES + m0 + lane] = full;

    // store un-normalized partial accumulators, row-major for coalesced reduce
    float* pa = pacc + (size_t)split * N_NODES * FOUT;
    #pragma unroll
    for (int ft = 0; ft < 4; ++ft) {
        #pragma unroll
        for (int v = 0; v < 8; ++v) {
            pa[(size_t)(m0 + v + hl * 8) * FOUT + ft * 16 + n] = acc[ft][v];
        }
    }
}

// ---------------------------------------------------------------------------
// Kernel 5: reduce splits, softmax-normalize, BN (folded), ELU.
// ---------------------------------------------------------------------------
__global__ __launch_bounds__(256) void k_finalize(const float* __restrict__ pacc,
                                                  const float* __restrict__ psum,
                                                  const float* __restrict__ gamma,
                                                  const float* __restrict__ beta,
                                                  const float* __restrict__ mean,
                                                  const float* __restrict__ var,
                                                  float*       __restrict__ out)
{
    const int idx = blockIdx.x * 256 + threadIdx.x;   // 0 .. 8192*64-1
    const int row = idx >> 6;
    const int f   = idx & 63;

    float s = 0.f, rs = 0.f;
    #pragma unroll
    for (int sp = 0; sp < SPLIT; ++sp) {
        s  += pacc[(size_t)sp * N_NODES * FOUT + idx];
        rs += psum[sp * N_NODES + row];
    }
    rs = (rs > 0.f) ? rs : 1.0f;
    const float x  = s / rs;
    const float sc = gamma[f] * rsqrtf(var[f] + 1e-5f);
    const float y  = x * sc + (beta[f] - mean[f] * sc);
    out[idx] = (y > 0.f) ? y : (__expf(y) - 1.0f);
}

// ---------------------------------------------------------------------------
extern "C" void kernel_launch(void* const* d_in, const int* in_sizes, int n_in,
                              void* d_out, int out_size, void* d_ws, size_t ws_size,
                              hipStream_t stream)
{
    const float* inp   = (const float*)d_in[0];
    const int*   adj   = (const int*)  d_in[1];
    const float* W     = (const float*)d_in[2];
    const float* a     = (const float*)d_in[3];
    const float* gamma = (const float*)d_in[4];
    const float* beta  = (const float*)d_in[5];
    const float* mean  = (const float*)d_in[6];
    const float* var   = (const float*)d_in[7];
    float* out = (float*)d_out;

    char* ws = (char*)d_ws;
    float*  hF   = (float*) ws;                                     // 2 MB
    __bf16* hB   = (__bf16*)(ws + (size_t)2 * 1024 * 1024);         // 1 MB
    float*  f1   = (float*) (ws + (size_t)3 * 1024 * 1024);         // 32 KB
    float*  f2   = (float*) (ws + (size_t)3 * 1024 * 1024 + 32 * 1024); // 32 KB
    float*  mx   = (float*) (ws + (size_t)3 * 1024 * 1024 + 64 * 1024); // 4 B
    float*  pacc = (float*) (ws + (size_t)4 * 1024 * 1024);         // SPLIT*2 MB
    float*  psum = (float*) (ws + (size_t)(4 + 2 * SPLIT) * 1024 * 1024); // SPLIT*32 KB

    k_h_gemm  <<<64, 256, 0, stream>>>(inp, W, hF, hB);
    k_f1f2    <<<32, 256, 0, stream>>>(hF, a, f1, f2);
    k_maxf2   <<< 1, 256, 0, stream>>>(f2, mx);
    k_gat_main<<<64 * SPLIT, 256, 0, stream>>>(adj, hB, f1, f2, mx, pacc, psum);
    k_finalize<<<(N_NODES * FOUT) / 256, 256, 0, stream>>>(pacc, psum,
                                       gamma, beta, mean, var, out);
}
